// Gate_25967372272135
// MI455X (gfx1250) — compile-verified, matrix operand order checked by reading, not run on hardware
//
#include <hip/hip_runtime.h>
#include <hip/hip_bf16.h>

typedef __attribute__((ext_vector_type(16))) __bf16 v16bf;
typedef __attribute__((ext_vector_type(8)))  __bf16 v8bf;
typedef __attribute__((ext_vector_type(4)))  __bf16 v4bf;
typedef __attribute__((ext_vector_type(8)))  float  v8f;
typedef __attribute__((ext_vector_type(4)))  float  v4f;

#define N_EXPERTS 256
#define N_GROUPS  8
#define TOPK_G    4
#define TOPK      8
#define DIM       7168
#define T_TOK     8192
#define TM        32        // tokens per workgroup
#define KSTEP     32
#define NKS       (DIM / KSTEP)   // 224
#define A_STRIDE  40        // bf16 elems per LDS row (32 + pad; 80B, conflict-free)
#define S_STRIDE  260       // score row stride (floats), avoids bank conflicts

// ---------------- kernel 0: split weight f32 -> bf16 hi/lo planes ----------------
__global__ __launch_bounds__(256) void wsplit_kernel(const float* __restrict__ w,
                                                     __bf16* __restrict__ whi,
                                                     __bf16* __restrict__ wlo) {
  size_t i = ((size_t)blockIdx.x * 256 + threadIdx.x) * 4;
  v4f v = *(const v4f*)(w + i);
  v4bf h, l;
#pragma unroll
  for (int k = 0; k < 4; ++k) {
    __bf16 hh = (__bf16)v[k];
    h[k] = hh;
    l[k] = (__bf16)(v[k] - (float)hh);
  }
  *(v4bf*)(whi + i) = h;
  *(v4bf*)(wlo + i) = l;
}

// ---------------- wave32 reductions ----------------
__device__ __forceinline__ void wave_argmax(float& v, int& idx) {
#pragma unroll
  for (int off = 16; off > 0; off >>= 1) {
    float ov = __shfl_xor(v, off, 32);
    int   oi = __shfl_xor(idx, off, 32);
    if (ov > v || (ov == v && oi < idx)) { v = ov; idx = oi; }
  }
}
__device__ __forceinline__ float wave_max(float v) {
#pragma unroll
  for (int off = 16; off > 0; off >>= 1) v = fmaxf(v, __shfl_xor(v, off, 32));
  return v;
}

__device__ __forceinline__ v16bf cat16(v8bf a, v8bf b) {
  return __builtin_shufflevector(a, b, 0,1,2,3,4,5,6,7,8,9,10,11,12,13,14,15);
}

// ---------------- kernel 1: fused scorer GEMM + sigmoid + grouped top-k ----------------
__global__ __launch_bounds__(256) void gate_kernel(const float* __restrict__ x,
                                                   const __bf16* __restrict__ whi,
                                                   const __bf16* __restrict__ wlo,
                                                   const float* __restrict__ bias,
                                                   float* __restrict__ outw,
                                                   int* __restrict__ outi) {
  __shared__ __align__(16) __bf16 sA_hi[2][TM][A_STRIDE];
  __shared__ __align__(16) __bf16 sA_lo[2][TM][A_STRIDE];
  __shared__ float sScore[TM][S_STRIDE];

  const int tid  = threadIdx.x;
  const int lane = tid & 31;
  const int wave = tid >> 5;       // 0..7 -> experts [32w, 32w+32)
  const int l15  = lane & 15;
  const int ksel = (lane >> 4) * 8;  // K-segment base per WMMA 16-bit layout

  // cooperative A loader mapping: 32 rows x 32 K, 4 f32 per thread
  const int arow = tid >> 3;
  const int akq  = (tid & 7) * 4;
  const float* xptr = x + ((size_t)blockIdx.x * TM + arow) * DIM + akq;

  auto cvtStore = [&](int buf, v4f v) {
    v4bf h, l;
#pragma unroll
    for (int k = 0; k < 4; ++k) {
      __bf16 hh = (__bf16)v[k];
      h[k] = hh;
      l[k] = (__bf16)(v[k] - (float)hh);
    }
    *(v4bf*)&sA_hi[buf][arow][akq] = h;
    *(v4bf*)&sA_lo[buf][arow][akq] = l;
  };

  // per-lane B column pointers (weight row e == B column e)
  const size_t e0 = (size_t)(wave * 32 + l15);
  const size_t e1 = e0 + 16;
  const __bf16* bh0p = whi + e0 * DIM + ksel;
  const __bf16* bl0p = wlo + e0 * DIM + ksel;
  const __bf16* bh1p = whi + e1 * DIM + ksel;
  const __bf16* bl1p = wlo + e1 * DIM + ksel;

  v8f c00 = {}, c01 = {}, c10 = {}, c11 = {};

  // prologue: stage K-step 0
  cvtStore(0, *(const v4f*)xptr);
  __syncthreads();

  for (int ks = 0; ks < NKS; ++ks) {
    const int buf = ks & 1;
    const bool more = (ks + 1) < NKS;
    v4f anext;
    if (more) anext = *(const v4f*)(xptr + (size_t)(ks + 1) * KSTEP);

    const int k0 = ks * KSTEP;

    // B tiles: bf16, per-lane contiguous 16B segments (L2-resident)
    v16bf bhi0 = cat16(*(const v8bf*)(bh0p + k0), *(const v8bf*)(bh0p + k0 + 16));
    v16bf blo0 = cat16(*(const v8bf*)(bl0p + k0), *(const v8bf*)(bl0p + k0 + 16));
    v16bf bhi1 = cat16(*(const v8bf*)(bh1p + k0), *(const v8bf*)(bh1p + k0 + 16));
    v16bf blo1 = cat16(*(const v8bf*)(bl1p + k0), *(const v8bf*)(bl1p + k0 + 16));

    // A tiles from LDS (ds_load_b128 x2 per tile)
    v16bf ahi0 = cat16(*(const v8bf*)&sA_hi[buf][l15][ksel],
                       *(const v8bf*)&sA_hi[buf][l15][ksel + 16]);
    v16bf alo0 = cat16(*(const v8bf*)&sA_lo[buf][l15][ksel],
                       *(const v8bf*)&sA_lo[buf][l15][ksel + 16]);
    v16bf ahi1 = cat16(*(const v8bf*)&sA_hi[buf][16 + l15][ksel],
                       *(const v8bf*)&sA_hi[buf][16 + l15][ksel + 16]);
    v16bf alo1 = cat16(*(const v8bf*)&sA_lo[buf][16 + l15][ksel],
                       *(const v8bf*)&sA_lo[buf][16 + l15][ksel + 16]);

    // 3-term bf16 decomposition: hi*hi + hi*lo + lo*hi  (~fp32-accurate)
    c00 = __builtin_amdgcn_wmma_f32_16x16x32_bf16(false, ahi0, false, bhi0, (short)0, c00, false, false);
    c00 = __builtin_amdgcn_wmma_f32_16x16x32_bf16(false, ahi0, false, blo0, (short)0, c00, false, false);
    c00 = __builtin_amdgcn_wmma_f32_16x16x32_bf16(false, alo0, false, bhi0, (short)0, c00, false, false);

    c01 = __builtin_amdgcn_wmma_f32_16x16x32_bf16(false, ahi0, false, bhi1, (short)0, c01, false, false);
    c01 = __builtin_amdgcn_wmma_f32_16x16x32_bf16(false, ahi0, false, blo1, (short)0, c01, false, false);
    c01 = __builtin_amdgcn_wmma_f32_16x16x32_bf16(false, alo0, false, bhi1, (short)0, c01, false, false);

    c10 = __builtin_amdgcn_wmma_f32_16x16x32_bf16(false, ahi1, false, bhi0, (short)0, c10, false, false);
    c10 = __builtin_amdgcn_wmma_f32_16x16x32_bf16(false, ahi1, false, blo0, (short)0, c10, false, false);
    c10 = __builtin_amdgcn_wmma_f32_16x16x32_bf16(false, alo1, false, bhi0, (short)0, c10, false, false);

    c11 = __builtin_amdgcn_wmma_f32_16x16x32_bf16(false, ahi1, false, bhi1, (short)0, c11, false, false);
    c11 = __builtin_amdgcn_wmma_f32_16x16x32_bf16(false, ahi1, false, blo1, (short)0, c11, false, false);
    c11 = __builtin_amdgcn_wmma_f32_16x16x32_bf16(false, alo1, false, bhi1, (short)0, c11, false, false);

    if (more) cvtStore(buf ^ 1, anext);
    __syncthreads();
  }

  // epilogue: sigmoid -> LDS score board. C layout: M = r + 8*(lane>>4), N = lane&15.
  auto dumpC = [&](v8f c, int mt, int ntl) {
    const int trow = mt * 16 + (lane >> 4) * 8;
    const int e = wave * 32 + ntl * 16 + l15;
#pragma unroll
    for (int r = 0; r < 8; ++r)
      sScore[trow + r][e] = 1.0f / (1.0f + __expf(-c[r]));
  };
  dumpC(c00, 0, 0); dumpC(c01, 0, 1); dumpC(c10, 1, 0); dumpC(c11, 1, 1);
  __syncthreads();

  // ---------------- routing: each wave handles 4 tokens ----------------
  float bvals[8];
#pragma unroll
  for (int j = 0; j < 8; ++j) bvals[j] = bias[j * 32 + lane];

  for (int i = 0; i < 4; ++i) {
    const int tl = wave + 8 * i;
    float sc[8], sv[8];
#pragma unroll
    for (int j = 0; j < 8; ++j) {         // lane owns experts e = j*32 + lane (group j)
      float s = sScore[tl][j * 32 + lane];
      sc[j] = s;
      sv[j] = s + bvals[j];
    }
    // group score = top-2 sum of biased scores within each 32-expert group
    float gs[8];
#pragma unroll
    for (int j = 0; j < 8; ++j) {
      float v = sv[j]; int ix = lane;
      wave_argmax(v, ix);
      float v2 = (lane == ix) ? -INFINITY : sv[j];
      gs[j] = v + wave_max(v2);
    }
    // top-4 groups (uniform across lanes)
    unsigned selmask = 0;
#pragma unroll
    for (int r = 0; r < TOPK_G; ++r) {
      float best = -INFINITY; int bg = 0;
#pragma unroll
      for (int j = 0; j < 8; ++j)
        if (!((selmask >> j) & 1) && gs[j] > best) { best = gs[j]; bg = j; }
      selmask |= 1u << bg;
    }
    float mv[8];
#pragma unroll
    for (int j = 0; j < 8; ++j) mv[j] = ((selmask >> j) & 1) ? sv[j] : -INFINITY;

    float wts[8]; int idxs[8]; float wsum = 0.0f;
#pragma unroll
    for (int r = 0; r < TOPK; ++r) {
      float lv = -INFINITY; int le = N_EXPERTS + lane;  // low-index tie-break
#pragma unroll
      for (int j = 0; j < 8; ++j) {
        int ce = j * 32 + lane;
        if (mv[j] > lv) { lv = mv[j]; le = ce; }
      }
      wave_argmax(lv, le);                // (value, expert) uniform on all lanes
      const int jj = le >> 5;
#pragma unroll
      for (int j = 0; j < 8; ++j)
        if (jj == j && (le & 31) == lane) mv[j] = -INFINITY;   // clear winner
      float wsel = sScore[tl][le];        // original sigmoid score
      wts[r] = wsel; idxs[r] = le; wsum += wsel;
    }
    if (lane == 0) {
      const size_t tg = (size_t)blockIdx.x * TM + tl;
      const float scale = 2.5f / wsum;
#pragma unroll
      for (int r = 0; r < TOPK; ++r) {
        outw[tg * TOPK + r] = wts[r] * scale;
        outi[tg * TOPK + r] = idxs[r];
      }
    }
  }
}

extern "C" void kernel_launch(void* const* d_in, const int* in_sizes, int n_in,
                              void* d_out, int out_size, void* d_ws, size_t ws_size,
                              hipStream_t stream) {
  const float* x    = (const float*)d_in[0];
  const float* w    = (const float*)d_in[1];
  const float* bias = (const float*)d_in[2];

  float* outw = (float*)d_out;
  int*   outi = (int*)((float*)d_out + (size_t)T_TOK * TOPK);   // indices after weights

  __bf16* whi = (__bf16*)d_ws;
  __bf16* wlo = whi + (size_t)N_EXPERTS * DIM;

  // kernel 0: split the 7.3 MB weight matrix into bf16 hi/lo planes (one-time, L2-hot after)
  wsplit_kernel<<<(N_EXPERTS * DIM) / (256 * 4), 256, 0, stream>>>(w, whi, wlo);

  // kernel 1: fused scorer GEMM (bf16 WMMA, 3-term) + sigmoid + grouped top-k routing
  gate_kernel<<<T_TOK / TM, 256, 0, stream>>>(x, whi, wlo, bias, outw, outi);
}
